// GCN_mamba_block_37263136260746
// MI455X (gfx1250) — matrix operand order repaired and checked by reading, not compiled, not run on hardware
//
#include <hip/hip_runtime.h>
#include <hip/hip_bf16.h>
#include <cstdint>
#include <cstddef>

// ---------------- types for WMMA (CDNA5 / gfx1250, wave32) ----------------
typedef __bf16 bf16;
typedef bf16  v16bf __attribute__((ext_vector_type(16)));
typedef bf16  v8bf  __attribute__((ext_vector_type(8)));
typedef bf16  v4bf  __attribute__((ext_vector_type(4)));
typedef float v8f   __attribute__((ext_vector_type(8)));

#define NN    8192      // nodes (and K of the adjacency GEMM)
#define DIN   256       // d_inner
#define DST   16        // d_state
#define DTR   16        // dt_rank
#define DMOD  256       // d_model
#define LHOP  8         // layers / SSM sequence length
#define ROWS  (NN * LHOP)   // 65536 rows of the (N,L) flattened tensors

// round-to-nearest-even f32 -> bf16
__device__ __forceinline__ bf16 f2bf(float f) {
  union { float f; uint32_t u; } in; in.f = f;
  uint32_t u = in.u;
  uint32_t r = (u + 0x7fffu + ((u >> 16) & 1u)) >> 16;
  union { uint16_t s; bf16 b; } out; out.s = (uint16_t)r;
  return out.b;
}

__device__ __forceinline__ v16bf pack16(v8bf lo, v8bf hi) {
  v16bf r;
#pragma unroll
  for (int i = 0; i < 8; ++i) { r[i] = lo[i]; r[i + 8] = hi[i]; }
  return r;
}

// ---------------- one-time conversion kernels ----------------
__global__ __launch_bounds__(256) void cvt_adj_kernel(const float4* __restrict__ a,
                                                      v4bf* __restrict__ o) {
  size_t tid = (size_t)blockIdx.x * 256 + threadIdx.x;
  size_t stride = (size_t)gridDim.x * 256;
  const size_t total = (size_t)NN * NN / 4;
  for (size_t i = tid; i < total; i += stride) {
    float4 v = a[i];
    v4bf b;
    b[0] = f2bf(v.x); b[1] = f2bf(v.y); b[2] = f2bf(v.z); b[3] = f2bf(v.w);
    o[i] = b;
  }
}

// u[:,0,:] = x ;  xiT0[n][m] = bf16(x[m][n])
__global__ __launch_bounds__(256) void init_x_kernel(const float* __restrict__ x,
                                                     float* __restrict__ U,
                                                     bf16* __restrict__ xiT0) {
  size_t tid = (size_t)blockIdx.x * 256 + threadIdx.x;   // 8192*256 threads, one elem each
  int m = (int)(tid >> 8);
  int n = (int)(tid & 255);
  float v = x[tid];
  U[(size_t)m * (LHOP * DIN) + n] = v;
  xiT0[(size_t)n * NN + m] = f2bf(v);
}

// WoutT[n][k] = bf16(W_out[k][n])
__global__ __launch_bounds__(256) void cvt_wout_kernel(const float* __restrict__ W,
                                                       bf16* __restrict__ WT) {
  int tid = blockIdx.x * 256 + threadIdx.x;              // 65536 threads
  int n = tid >> 8, k = tid & 255;
  WT[tid] = f2bf(W[k * 256 + n]);
}

// ---------------- hop GEMM: xi_new = 0.95*(adj@xi) + 0.05*x ----------------
// A  = adjB  row-major [8192 x 8192] bf16
// BT = xiT   row-major [256 x 8192]  bf16  (BT[n][k] = xi[k][n])
// WG = 8 waves, wave tile 32(M) x 64(N): 2 M-frags x 4 N-frags = 8 WMMAs / 32-K step.
// WG tile 128(M) x 128(N); grid (64, 2).
__device__ __forceinline__ void epilogue_hop(const v8f& acc, int m0, int n_g,
                                             const float* __restrict__ x0,
                                             float* __restrict__ u_hop,
                                             bf16* __restrict__ xiTn) {
  v8bf xr;
#pragma unroll
  for (int j = 0; j < 8; ++j) {
    float val = 0.95f * acc[j] + 0.05f * x0[(size_t)(m0 + j) * DIN + n_g];
    u_hop[(size_t)(m0 + j) * (LHOP * DIN) + n_g] = val;
    xr[j] = f2bf(val);
  }
  *(v8bf*)(xiTn + (size_t)n_g * NN + m0) = xr;   // 16B contiguous store
}

__global__ __launch_bounds__(256) void gemm_hop_kernel(const bf16* __restrict__ A,
                                                       const bf16* __restrict__ BT,
                                                       const float* __restrict__ x0,
                                                       float* __restrict__ u_hop,
                                                       bf16* __restrict__ xiTn) {
  const int K = NN;
  const int lane = threadIdx.x & 31, wv = threadIdx.x >> 5;
  const int half = lane >> 4, r = lane & 15;
  const int mw = wv & 3, nw = wv >> 2;                 // 4 waves in M, 2 in N
  const int m_base = blockIdx.x * 128 + mw * 32;
  const int n_base = blockIdx.y * 128 + nw * 64;

  // A fragment: lane(<16) row M=r holds K=0..15 contiguous; lanes>=16 K=16..31 (matches B)
  const bf16* ap0 = A + (size_t)(m_base + r) * K + half * 16;
  const bf16* ap1 = ap0 + (size_t)16 * K;
  // B fragment (same contiguous k order as A): lane(<16) col N=r holds K=0..15 contiguous; lanes>=16 K=16..31
  const bf16* bp0 = BT + (size_t)(n_base + r) * K + half * 16;
  const bf16* bp1 = bp0 + (size_t)16 * K;
  const bf16* bp2 = bp0 + (size_t)32 * K;
  const bf16* bp3 = bp0 + (size_t)48 * K;

  v8f acc[2][4];
#pragma unroll
  for (int i = 0; i < 2; ++i)
#pragma unroll
    for (int t = 0; t < 4; ++t) acc[i][t] = v8f{};

  for (int k0 = 0; k0 < K; k0 += 128) {
    // one prefetch batch per 32 WMMAs; +1024 elems (=2KB) ahead -> GL2 warm-up for streamed rows
    __builtin_prefetch(ap0 + k0 + 1024, 0, 0);
    __builtin_prefetch(ap1 + k0 + 1024, 0, 0);
    __builtin_prefetch(bp0 + k0 + 1024, 0, 0);
    __builtin_prefetch(bp1 + k0 + 1024, 0, 0);
    __builtin_prefetch(bp2 + k0 + 1024, 0, 0);
    __builtin_prefetch(bp3 + k0 + 1024, 0, 0);
#pragma unroll
    for (int kk = 0; kk < 128; kk += 32) {
      const int k = k0 + kk;
      v16bf a[2], b[4];
      a[0] = *(const v16bf*)(ap0 + k);
      a[1] = *(const v16bf*)(ap1 + k);
      b[0] = *(const v16bf*)(bp0 + k);
      b[1] = *(const v16bf*)(bp1 + k);
      b[2] = *(const v16bf*)(bp2 + k);
      b[3] = *(const v16bf*)(bp3 + k);
#pragma unroll
      for (int i = 0; i < 2; ++i)
#pragma unroll
        for (int t = 0; t < 4; ++t)
          acc[i][t] = __builtin_amdgcn_wmma_f32_16x16x32_bf16(
              false, a[i], false, b[t], (short)0, acc[i][t], false, false);
    }
  }

  // C/D layout: lanes 0-15: N=lane, VGPR j -> M=j ; lanes 16-31: N=lane-16, M=j+8
#pragma unroll
  for (int i = 0; i < 2; ++i)
#pragma unroll
    for (int t = 0; t < 4; ++t)
      epilogue_hop(acc[i][t], m_base + i * 16 + half * 8, n_base + t * 16 + r,
                   x0, u_hop, xiTn);
}

// ---------------- fused selective-scan (+ input projections) ----------------
// One workgroup per node; thread = channel d (0..255); h[16] kept in VGPRs.
__global__ __launch_bounds__(256) void scan_kernel(const float* __restrict__ U,
                                                   const float* __restrict__ Wx,   // [256][48]
                                                   const float* __restrict__ Wdt,  // [16][256]
                                                   const float* __restrict__ A_log,// [256][16]
                                                   const float* __restrict__ Dv,   // [256]
                                                   bf16* __restrict__ yB) {
  __shared__ float sU[DIN];
  __shared__ float sX[DTR + 2 * DST];   // 48
  const int n = blockIdx.x;
  const int d = threadIdx.x;

  float h[DST], Ar[DST];
#pragma unroll
  for (int s = 0; s < DST; ++s) { h[s] = 0.f; Ar[s] = -__expf(A_log[d * DST + s]); }
  const float Dd = Dv[d];
  const float* urow = U + (size_t)n * (LHOP * DIN);

  for (int l = 0; l < LHOP; ++l) {
    float u_ld = urow[l * DIN + d];
    sU[d] = u_ld;
    __syncthreads();
    if (d < DTR + 2 * DST) {                    // x_dbl = relu(u @ W_xproj)
      float s_ = 0.f;
      for (int k = 0; k < DIN; ++k) s_ = fmaf(sU[k], Wx[k * 48 + d], s_);
      sX[d] = fmaxf(s_, 0.f);
    }
    __syncthreads();
    float dsum = 0.f;                           // delta = softplus(x_dbl[:16] @ W_dtproj)
#pragma unroll
    for (int t = 0; t < DTR; ++t) dsum = fmaf(sX[t], Wdt[t * DIN + d], dsum);
    float delta = (dsum > 20.f) ? dsum : __logf(1.f + __expf(dsum));
    float y = 0.f;
#pragma unroll
    for (int s = 0; s < DST; ++s) {
      float dA = __expf(delta * Ar[s]);                       // deltaA
      h[s] = fmaf(dA, h[s], delta * sX[DTR + s] * u_ld);      // h = dA*h + delta*B*u
      y = fmaf(h[s], sX[DTR + DST + s], y);                   // y += h*C
    }
    y = fmaf(u_ld, Dd, y);                                    // + u*D
    yB[((size_t)n * LHOP + l) * DIN + d] = f2bf(y);
    __syncthreads();
  }
}

// ---------------- out-projection GEMM: out = y @ W_outproj ----------------
// M = 65536 so keep the grid wide: WG tile 128x128, wave tile 32x64, grid (512, 2).
__global__ __launch_bounds__(256) void outproj_kernel(const bf16* __restrict__ A,   // [65536][256]
                                                      const bf16* __restrict__ BT,  // [256][256]
                                                      float* __restrict__ out) {
  const int K = DIN;
  const int lane = threadIdx.x & 31, wv = threadIdx.x >> 5;
  const int half = lane >> 4, r = lane & 15;
  const int mw = wv & 3, nw = wv >> 2;
  const int m_base = blockIdx.x * 128 + mw * 32;
  const int n_base = blockIdx.y * 128 + nw * 64;

  const bf16* ap0 = A + (size_t)(m_base + r) * K + half * 16;
  const bf16* ap1 = ap0 + (size_t)16 * K;
  const bf16* bp0 = BT + (size_t)(n_base + r) * K + half * 16;
  const bf16* bp1 = bp0 + (size_t)16 * K;
  const bf16* bp2 = bp0 + (size_t)32 * K;
  const bf16* bp3 = bp0 + (size_t)48 * K;

  v8f acc[2][4];
#pragma unroll
  for (int i = 0; i < 2; ++i)
#pragma unroll
    for (int t = 0; t < 4; ++t) acc[i][t] = v8f{};

#pragma unroll
  for (int k = 0; k < K; k += 32) {
    v16bf a[2], b[4];
    a[0] = *(const v16bf*)(ap0 + k);
    a[1] = *(const v16bf*)(ap1 + k);
    b[0] = *(const v16bf*)(bp0 + k);
    b[1] = *(const v16bf*)(bp1 + k);
    b[2] = *(const v16bf*)(bp2 + k);
    b[3] = *(const v16bf*)(bp3 + k);
#pragma unroll
    for (int i = 0; i < 2; ++i)
#pragma unroll
      for (int t = 0; t < 4; ++t)
        acc[i][t] = __builtin_amdgcn_wmma_f32_16x16x32_bf16(
            false, a[i], false, b[t], (short)0, acc[i][t], false, false);
  }

#pragma unroll
  for (int i = 0; i < 2; ++i)
#pragma unroll
    for (int t = 0; t < 4; ++t) {
      const int m0 = m_base + i * 16 + half * 8;
      const int n_g = n_base + t * 16 + r;
#pragma unroll
      for (int j = 0; j < 8; ++j)
        out[(size_t)(m0 + j) * DMOD + n_g] = acc[i][t][j];
    }
}

// ---------------- launch ----------------
extern "C" void kernel_launch(void* const* d_in, const int* in_sizes, int n_in,
                              void* d_out, int out_size, void* d_ws, size_t ws_size,
                              hipStream_t stream) {
  (void)in_sizes; (void)n_in; (void)out_size; (void)ws_size;
  const float* x     = (const float*)d_in[0];
  const float* adj   = (const float*)d_in[1];
  const float* Wx    = (const float*)d_in[2];
  const float* Wdt   = (const float*)d_in[3];
  const float* Wout  = (const float*)d_in[4];
  const float* A_log = (const float*)d_in[5];
  const float* Dv    = (const float*)d_in[6];

  char* ws = (char*)d_ws;
  // workspace layout
  bf16*  adjB  = (bf16*)ws;                                            // 128 MB
  bf16*  xiT0  = (bf16*)(ws + (size_t)134217728);                      // 4 MB
  bf16*  xiT1  = xiT0 + (size_t)DIN * NN;                              // 4 MB
  float* U     = (float*)(ws + (size_t)134217728 + 2u * 4194304u);     // 64 MB  [N][L][256]
  bf16*  yB    = (bf16*)((char*)U + (size_t)67108864);                 // 32 MB  [N*L][256]
  bf16*  WoutT = (bf16*)((char*)yB + (size_t)33554432);                // 128 KB

  cvt_adj_kernel<<<8192, 256, 0, stream>>>((const float4*)adj, (v4bf*)adjB);
  init_x_kernel<<<8192, 256, 0, stream>>>(x, U, xiT0);
  cvt_wout_kernel<<<256, 256, 0, stream>>>(Wout, WoutT);

  bf16* cur = xiT0;
  bf16* nxt = xiT1;
  for (int hop = 1; hop < LHOP; ++hop) {
    gemm_hop_kernel<<<dim3(NN / 128, DIN / 128), 256, 0, stream>>>(
        adjB, cur, x, U + (size_t)hop * DIN, nxt);
    bf16* t = cur; cur = nxt; nxt = t;
  }

  scan_kernel<<<NN, 256, 0, stream>>>(U, Wx, Wdt, A_log, Dv, yB);

  outproj_kernel<<<dim3(ROWS / 128, DMOD / 128), 256, 0, stream>>>(yB, WoutT, (float*)d_out);
}
